// Joiner_79328045957280
// MI455X (gfx1250) — matrix operand order repaired and checked by reading, not compiled, not run on hardware
//
#include <hip/hip_runtime.h>
#include <hip/hip_bf16.h>

typedef __attribute__((ext_vector_type(16))) _Float16 v16h;
typedef __attribute__((ext_vector_type(8)))  _Float16 v8h;
typedef __attribute__((ext_vector_type(8)))  float    v8f;
typedef __attribute__((ext_vector_type(4)))  float    v4f;

#define N_   8
#define T_   512
#define U_   64
#define J_   512
#define V_   500
#define VPAD 512

__device__ __forceinline__ v16h join16(v8h lo, v8h hi) {
  v16h r;
#pragma unroll
  for (int i = 0; i < 8; ++i) { r[i] = lo[i]; r[i + 8] = hi[i]; }
  return r;
}

// Fast tanh via hardware TRANS ops: tanh(x) = 1 - 2/(exp(2x)+1).
// v_exp_f32 + v_rcp_f32; exact at saturation (-1/+1), error << f16 ulp.
__device__ __forceinline__ float fast_tanh(float x) {
  float e = __builtin_amdgcn_exp2f(x * 2.8853900817779268f); // 2*log2(e)
  return 1.0f - 2.0f * __builtin_amdgcn_rcpf(e + 1.0f);
}

// ---------------- convert kernels ----------------

__global__ void cvt_f32_f16(const float* __restrict__ src,
                            _Float16* __restrict__ dst, int n) {
  int i = blockIdx.x * blockDim.x + threadIdx.x;
  if (i < n) dst[i] = (_Float16)src[i];
}

// out_W (500x512 f32) -> B-fragment-contiguous f16 layout:
// dst[((vt*16 + kb)*32 + lane)*16 + h] = out_W[v][j],
//   v = vt*16 + (lane&15),  j = kb*32 + (lane>>4)*16 + h   (v>=500 padded 0)
__global__ void swizzle_outw(const float* __restrict__ src,
                             _Float16* __restrict__ dst) {
  int i = blockIdx.x * blockDim.x + threadIdx.x; // 0 .. 512*512-1
  int h    = i & 15;
  int lane = (i >> 4) & 31;
  int kb   = (i >> 9) & 15;
  int vt   = i >> 13;
  int v = vt * 16 + (lane & 15);
  int j = kb * 32 + (lane >> 4) * 16 + h;
  dst[i] = (v < V_) ? (_Float16)src[v * J_ + j] : (_Float16)0.0f;
}

// ---------------- projection GEMM: Y[rows x 512] = X * W^T + b ----------------

__global__ __launch_bounds__(256) void proj_wmma(
    const _Float16* __restrict__ X,
    const _Float16* __restrict__ W,
    const float* __restrict__ bias,
    float* __restrict__ Y,
    int rowTiles) {
  int wave = threadIdx.x >> 5;
  int lane = threadIdx.x & 31;
  int tile = blockIdx.x * 8 + wave;
  int rt = tile >> 5;
  int jt = tile & 31;
  if (rt >= rowTiles) return;

  int g  = lane >> 4;
  int mr = lane & 15;

  const _Float16* xrow = X + (size_t)(rt * 16 + mr) * J_;
  const _Float16* wrow = W + (size_t)(jt * 16 + mr) * J_ + g * 16;

  v8f acc = {};
#pragma unroll 4
  for (int kb = 0; kb < 16; ++kb) {
    int ka = kb * 32 + g * 16;
    v8h alo = *(const v8h*)(xrow + ka);
    v8h ahi = *(const v8h*)(xrow + ka + 8);
    v16h bb = *(const v16h*)(wrow + kb * 32);   // 16 contiguous halves, 32B aligned
    acc = __builtin_amdgcn_wmma_f32_16x16x32_f16(
        false, join16(alo, ahi), false, bb, (short)0, acc, false, false);
  }

  int j = jt * 16 + mr;
  float bj = bias[j];
#pragma unroll
  for (int r = 0; r < 8; ++r) {
    int m = r + 8 * g;
    Y[(size_t)(rt * 16 + m) * J_ + j] = acc[r] + bj;
  }
}

// ---------------- fused tanh + output GEMM ----------------
// One workgroup per (n, t): A tile = 64 u-rows x 512 k in LDS as WMMA
// A-fragments (64KB). 8 waves; wave handles 4 V-tiles; per K-step one B
// fragment is reused across 4 M-tiles (4 accumulators).

__global__ __launch_bounds__(256) void joiner_main(
    const float* __restrict__ EncP,    // (N*T) x 512 f32
    const float* __restrict__ DecP,    // (N*U) x 512 f32
    const _Float16* __restrict__ WfSw, // fragment-contiguous out_W f16
    const float* __restrict__ outB,    // 500 f32
    float* __restrict__ out) {         // N*T*U*500 f32
  __shared__ __align__(32) _Float16 sh[4 * 16 * 32 * 16]; // [mt][kb][lane][h] 64KB

  int b = blockIdx.x;        // n*T + t
  int n = b >> 9;
  int tid = threadIdx.x;

  const float* encRow  = EncP + (size_t)b * J_;
  const float* decBase = DecP + (size_t)(n * U_) * J_;

  // Phase 1: act[m][k] = tanh(enc[k]+dec[m][k]); 8 consecutive k per thread
  // -> one ds_store_b128. 64 rows x 64 groups = 4096 groups = 256 thr x 16.
#pragma unroll 4
  for (int it = 0; it < 16; ++it) {
    int gi = tid + it * 256;
    int k8 = gi & 63;        // which 8-wide k group
    int m  = gi >> 6;        // u row 0..63
    int k  = k8 * 8;
    const float* ep = encRow + k;
    const float* dp = decBase + (size_t)m * J_ + k;
    v4f e0 = *(const v4f*)ep;
    v4f e1 = *(const v4f*)(ep + 4);
    v4f d0 = *(const v4f*)dp;
    v4f d1 = *(const v4f*)(dp + 4);
    v8h hv;
#pragma unroll
    for (int e = 0; e < 4; ++e) {
      hv[e]     = (_Float16)fast_tanh(e0[e] + d0[e]);
      hv[e + 4] = (_Float16)fast_tanh(e1[e] + d1[e]);
    }
    int mt = m >> 4;
    int mr = m & 15;
    int kb = k8 >> 2;
    int lane = mr + (((k8 >> 1) & 1) << 4);
    int hb   = (k8 & 1) << 3;
    *(v8h*)&sh[(((mt * 16 + kb) * 32 + lane) * 16) + hb] = hv;
  }
  __syncthreads();

  int wave = tid >> 5;
  int lane = tid & 31;
  int g    = lane >> 4;
  int ncol = lane & 15;

  for (int i = 0; i < 4; ++i) {
    int vt = wave * 4 + i;
    // B fragment base: fragment id = vt*16+kb, 512 halves per fragment
    const _Float16* wp = WfSw + (size_t)(vt * 16) * 512 + lane * 16;
    v8f acc[4] = {};
#pragma unroll 4
    for (int kb = 0; kb < 16; ++kb) {
      v16h bb = *(const v16h*)(wp + kb * 512);            // coalesced 1KB/wave
      __builtin_prefetch(wp + (kb + 1) * 512, 0, 1);      // global_prefetch_b8
      const _Float16* ap = &sh[(size_t)(kb * 32 + lane) * 16];
#pragma unroll
      for (int mt = 0; mt < 4; ++mt) {
        v16h aa = *(const v16h*)(ap + mt * 8192);         // mt stride 16*32*16
        acc[mt] = __builtin_amdgcn_wmma_f32_16x16x32_f16(
            false, aa, false, bb, (short)0, acc[mt], false, false);
      }
    }
    int v = vt * 16 + ncol;
    if (v < V_) {
      float bv = outB[v];
#pragma unroll
      for (int mt = 0; mt < 4; ++mt) {
#pragma unroll
        for (int r = 0; r < 8; ++r) {
          int m = mt * 16 + r + 8 * g;
          out[((size_t)b * U_ + m) * V_ + v] = acc[mt][r] + bv;
        }
      }
    }
  }
}

// ---------------- launch ----------------

extern "C" void kernel_launch(void* const* d_in, const int* in_sizes, int n_in,
                              void* d_out, int out_size, void* d_ws, size_t ws_size,
                              hipStream_t stream) {
  (void)in_sizes; (void)n_in; (void)out_size; (void)ws_size;

  const float* encoder_out = (const float*)d_in[0];
  const float* decoder_out = (const float*)d_in[1];
  const float* enc_W = (const float*)d_in[2];
  const float* enc_b = (const float*)d_in[3];
  const float* dec_W = (const float*)d_in[4];
  const float* dec_b = (const float*)d_in[5];
  const float* out_W = (const float*)d_in[6];
  const float* out_b = (const float*)d_in[7];
  float* out = (float*)d_out;

  char* ws = (char*)d_ws;
  size_t o = 0;
  auto alloc = [&](size_t bytes) -> void* {
    void* p = ws + o;
    o = (o + bytes + 255) & ~(size_t)255;
    return p;
  };
  float*    EncP   = (float*)alloc((size_t)N_ * T_ * J_ * 4);
  float*    DecP   = (float*)alloc((size_t)N_ * U_ * J_ * 4);
  _Float16* enc16  = (_Float16*)alloc((size_t)N_ * T_ * J_ * 2);
  _Float16* dec16  = (_Float16*)alloc((size_t)N_ * U_ * J_ * 2);
  _Float16* encW16 = (_Float16*)alloc((size_t)J_ * J_ * 2);
  _Float16* decW16 = (_Float16*)alloc((size_t)J_ * J_ * 2);
  _Float16* WfSw   = (_Float16*)alloc((size_t)VPAD * J_ * 2);

  const int cvtB = 256;
  int nEnc = N_ * T_ * J_;
  int nDec = N_ * U_ * J_;
  int nW   = J_ * J_;
  int nWf  = VPAD * J_;

  cvt_f32_f16<<<(nEnc + cvtB - 1) / cvtB, cvtB, 0, stream>>>(encoder_out, enc16, nEnc);
  cvt_f32_f16<<<(nDec + cvtB - 1) / cvtB, cvtB, 0, stream>>>(decoder_out, dec16, nDec);
  cvt_f32_f16<<<(nW   + cvtB - 1) / cvtB, cvtB, 0, stream>>>(enc_W, encW16, nW);
  cvt_f32_f16<<<(nW   + cvtB - 1) / cvtB, cvtB, 0, stream>>>(dec_W, decW16, nW);
  swizzle_outw<<<(nWf + cvtB - 1) / cvtB, cvtB, 0, stream>>>(out_W, WfSw);

  proj_wmma<<<8192 / 8, 256, 0, stream>>>(enc16, encW16, enc_b, EncP, 256);
  proj_wmma<<<1024 / 8, 256, 0, stream>>>(dec16, decW16, dec_b, DecP, 32);

  // fused tanh + output GEMM: one workgroup per (n, t)
  joiner_main<<<N_ * T_, 256, 0, stream>>>(EncP, DecP, WfSw, out_b, out);
}
